// Proj_38800734552551
// MI455X (gfx1250) — matrix operand order repaired and checked by reading, not compiled, not run on hardware
//
#include <hip/hip_runtime.h>

// ---------------------------------------------------------------------------
// Masked-BN MLP, (B=8, C=256, N=32768).  x (B,C,N) is already the transposed
// activation matrix, so both layers are H = W @ X per batch, keeping (B,C,N)
// layout throughout.  GEMMs run in bf16 WMMA (16x16x32, f32 accum); masked BN
// statistics fused into GEMM epilogues; W panels staged to LDS with
// GLOBAL_LOAD_ASYNC_TO_LDS_B128 (ASYNCcnt).  Bandwidth-bound: ~69 GFLOP vs
// ~940 MB HBM traffic @ 23.3 TB/s.
// ---------------------------------------------------------------------------

#define E_   256
#define B_   8
#define N_   32768
#define M_   (B_ * N_)          // 262144 columns total
#define EPS_ 1e-5f

#define ETILE 64                // channels per block
#define NTILE 32                // columns per block
#define KPAD  264               // 256 + 8 halves; row = 528B = 33*16B (aligned, conflict-free)

typedef __bf16 bf16_t;
typedef __attribute__((ext_vector_type(16))) __bf16 v16bf;
typedef __attribute__((ext_vector_type(8)))  __bf16 v8bf;
typedef __attribute__((ext_vector_type(4)))  __bf16 v4bf;
typedef __attribute__((ext_vector_type(8)))  float  v8f;
typedef __attribute__((ext_vector_type(4)))  int    v4i;
typedef __attribute__((address_space(1)))    v4i    g_v4i;   // global AS pointer elem
typedef __attribute__((address_space(3)))    v4i    l_v4i;   // LDS AS pointer elem

// ---------------------------------------------------------------- helpers
__device__ __forceinline__ v16bf cat16(v8bf lo, v8bf hi) {
    return __builtin_shufflevector(lo, hi, 0,1,2,3,4,5,6,7,8,9,10,11,12,13,14,15);
}

__device__ __forceinline__ v8f wmma_bf16(v16bf a, v16bf b, v8f c) {
#if defined(__gfx1250__)
    return __builtin_amdgcn_wmma_f32_16x16x32_bf16(false, a, false, b, (short)0, c,
                                                   false, false);
#else
    c[0] += (float)a[0] * (float)b[0];   // host-pass placeholder, never executed
    return c;
#endif
}

// Async global->LDS copy of one 16B chunk (ASYNCcnt tracked).  Builtin takes
// (AS1 v4i* src, AS3 v4i* dst, imm offset, imm cpol); AS3 pointers are 32-bit
// so the integer cast naturally truncates a generic LDS address to its offset.
__device__ __forceinline__ void async_copy_b128(void* lds_dst, const void* gsrc) {
#if __has_builtin(__builtin_amdgcn_global_load_async_to_lds_b128)
    __builtin_amdgcn_global_load_async_to_lds_b128(
        (g_v4i*)(uintptr_t)gsrc,
        (l_v4i*)(uintptr_t)(uint32_t)(uintptr_t)lds_dst,
        0, 0);
#else
    uint32_t loff = (uint32_t)(uintptr_t)lds_dst;
    asm volatile("global_load_async_to_lds_b128 %0, %1, off"
                 :: "v"(loff), "v"(gsrc) : "memory");
#endif
}

__device__ __forceinline__ void wait_async0(void) {
#if __has_builtin(__builtin_amdgcn_s_wait_asynccnt)
    __builtin_amdgcn_s_wait_asynccnt(0);
#else
    asm volatile("s_wait_asynccnt 0x0" ::: "memory");
#endif
}

// ------------------------------------------------------- small utility kernels
__global__ __launch_bounds__(256) void k_convert_w(const float* __restrict__ w1,
                                                   const float* __restrict__ w2,
                                                   bf16_t* __restrict__ o1,
                                                   bf16_t* __restrict__ o2) {
    int i = blockIdx.x * 256 + threadIdx.x;      // 65536 total
    o1[i] = (bf16_t)w1[i];
    o2[i] = (bf16_t)w2[i];
}

__global__ __launch_bounds__(256) void k_count(const unsigned char* __restrict__ mask,
                                               float* __restrict__ cnt) {
    int i = blockIdx.x * 256 + threadIdx.x;      // 262144 total, no tail
    unsigned long long bal = __ballot(mask[i] != 0);
    if ((threadIdx.x & 31) == 0) atomicAdd(cnt, (float)__popcll(bal));
}

__global__ __launch_bounds__(256) void k_finalize(const float* __restrict__ sum,
                                                  const float* __restrict__ sq,
                                                  const float* __restrict__ cnt,
                                                  const float* __restrict__ g,
                                                  const float* __restrict__ bt,
                                                  float* __restrict__ scale,
                                                  float* __restrict__ shift) {
    int e = threadIdx.x;
    float n   = fmaxf(cnt[0], 1.0f);
    float mu  = sum[e] / n;
    float var = fmaxf(sq[e] / n - mu * mu, 0.0f);
    float sc  = g[e] * rsqrtf(var + EPS_);
    scale[e] = sc;
    shift[e] = bt[e] - mu * sc;
}

// ------------------------------------------------------------- GEMM core bits
// A fragment (16x32 bf16): lane<16 holds K {k0..k0+7, k0+16..k0+23} of row l16;
// lane>=16 holds K {k0+8..15, k0+24..31}.  B fragment (32x16): lane holds the
// same K split as A, for column l16.
__device__ __forceinline__ void wave_gemm(const bf16_t* __restrict__ sW,
                                          const bf16_t* __restrict__ sX,
                                          int et, int nt, int half, int l16,
                                          v8f& acc) {
    const bf16_t* wrow = sW + (et * 16 + l16) * KPAD + half * 8;
    const bf16_t* xrow = sX + (nt * 16 + l16) * KPAD + half * 8;
#pragma unroll
    for (int ks = 0; ks < 8; ++ks) {
        int k0 = ks * 32;
        v8bf alo = *(const v8bf*)(wrow + k0);
        v8bf ahi = *(const v8bf*)(wrow + k0 + 16);
        v8bf blo = *(const v8bf*)(xrow + k0);
        v8bf bhi = *(const v8bf*)(xrow + k0 + 16);
        acc = wmma_bf16(cat16(alo, ahi), cat16(blo, bhi), acc);
    }
}

// Async W panel stage: 64 rows x 256 bf16 = 2048 16B chunks over 256 threads.
__device__ __forceinline__ void stage_w_async(bf16_t* sW, const bf16_t* wpanel, int t) {
#pragma unroll
    for (int i = 0; i < 8; ++i) {
        int chunk = t + i * 256;
        int er = chunk >> 5, cc = chunk & 31;
        async_copy_b128(&sW[er * KPAD + cc * 8], wpanel + er * 256 + cc * 8);
    }
}

// ----------------------------------------------------------------- GEMM1
// H1[b] = W1 @ X[b] + b1 ; fused masked stats on h1.
__global__ __launch_bounds__(256)
void k_gemm1(const float* __restrict__ x, const unsigned char* __restrict__ mask,
             const bf16_t* __restrict__ w1b, const float* __restrict__ b1,
             bf16_t* __restrict__ h1, float* __restrict__ sum, float* __restrict__ sq) {
    __shared__ __align__(16) bf16_t sW[ETILE * KPAD];
    __shared__ __align__(16) bf16_t sX[NTILE * KPAD];

    const int t     = threadIdx.x;
    const int mbase = blockIdx.x * NTILE;        // global column base
    const int b     = mbase >> 15;               // batch
    const int n0    = mbase & (N_ - 1);
    const int eg    = blockIdx.y * ETILE;

    stage_w_async(sW, w1b + (size_t)eg * E_, t);

    // X tile: 32 cols x 256 chans, fp32 -> bf16, stored [n][c]
    {
        int n = t & 31, c0 = t >> 5;
#pragma unroll
        for (int i = 0; i < 32; ++i) {
            int c = c0 + i * 8;
            float v = x[(size_t)((b << 8) + c) * N_ + n0 + n];
            sX[n * KPAD + c] = (bf16_t)v;
        }
    }
    wait_async0();
    __syncthreads();

    const int wave = t >> 5, lane = t & 31;
    const int half = lane >> 4, l16 = lane & 15;
    const int et = wave >> 1, nt = wave & 1;

    v8f acc = {0.f, 0.f, 0.f, 0.f, 0.f, 0.f, 0.f, 0.f};
    wave_gemm(sW, sX, et, nt, half, l16, acc);

    const int ebase = eg + et * 16 + half * 8;
    const int col   = n0 + nt * 16 + l16;
    const float mv  = mask[mbase + nt * 16 + l16] ? 1.f : 0.f;
#pragma unroll
    for (int r = 0; r < 8; ++r) {
        int e = ebase + r;
        float v = acc[r] + b1[e];
        h1[(size_t)((b << 8) + e) * N_ + col] = (bf16_t)v;
        float s = v * mv, s2 = v * v * mv;
#pragma unroll
        for (int o = 1; o < 16; o <<= 1) { s += __shfl_xor(s, o); s2 += __shfl_xor(s2, o); }
        if (l16 == 0) { atomicAdd(&sum[e], s); atomicAdd(&sq[e], s2); }
    }
}

// ----------------------------------------------------------------- GEMM2
// H2[b] = W2 @ relu(bn1(H1[b])) + b2 ; bn1 fused into B staging; fused stats;
// H2 stored bf16 (halves the HBM traffic of the final passes).
__global__ __launch_bounds__(256)
void k_gemm2(const bf16_t* __restrict__ h1, const unsigned char* __restrict__ mask,
             const bf16_t* __restrict__ w2b, const float* __restrict__ b2,
             const float* __restrict__ scale1, const float* __restrict__ shift1,
             bf16_t* __restrict__ h2, float* __restrict__ sum, float* __restrict__ sq) {
    __shared__ __align__(16) bf16_t sW[ETILE * KPAD];
    __shared__ __align__(16) bf16_t sX[NTILE * KPAD];

    const int t     = threadIdx.x;
    const int mbase = blockIdx.x * NTILE;
    const int b     = mbase >> 15;
    const int n0    = mbase & (N_ - 1);
    const int eg    = blockIdx.y * ETILE;

    stage_w_async(sW, w2b + (size_t)eg * E_, t);

    {
        int n = t & 31, c0 = t >> 5;
#pragma unroll
        for (int i = 0; i < 32; ++i) {
            int c = c0 + i * 8;
            float v = (float)h1[(size_t)((b << 8) + c) * N_ + n0 + n];
            v = fmaxf(v * scale1[c] + shift1[c], 0.f);       // BN1 + ReLU fused
            sX[n * KPAD + c] = (bf16_t)v;
        }
    }
    wait_async0();
    __syncthreads();

    const int wave = t >> 5, lane = t & 31;
    const int half = lane >> 4, l16 = lane & 15;
    const int et = wave >> 1, nt = wave & 1;

    v8f acc = {0.f, 0.f, 0.f, 0.f, 0.f, 0.f, 0.f, 0.f};
    wave_gemm(sW, sX, et, nt, half, l16, acc);

    const int ebase = eg + et * 16 + half * 8;
    const int col   = n0 + nt * 16 + l16;
    const float mv  = mask[mbase + nt * 16 + l16] ? 1.f : 0.f;
#pragma unroll
    for (int r = 0; r < 8; ++r) {
        int e = ebase + r;
        float v = acc[r] + b2[e];
        h2[(size_t)((b << 8) + e) * N_ + col] = (bf16_t)v;
        float s = v * mv, s2 = v * v * mv;
#pragma unroll
        for (int o = 1; o < 16; o <<= 1) { s += __shfl_xor(s, o); s2 += __shfl_xor(s2, o); }
        if (l16 == 0) { atomicAdd(&sum[e], s); atomicAdd(&sq[e], s2); }
    }
}

// ----------------------------------------------------------------- output
__global__ __launch_bounds__(256)
void k_out(const bf16_t* __restrict__ h2, const unsigned char* __restrict__ mask,
           const float* __restrict__ scale2, const float* __restrict__ shift2,
           float* __restrict__ out) {
    size_t i4 = ((size_t)blockIdx.x * 256 + threadIdx.x) * 4;   // 67108864 total
    int e = (int)((i4 >> 15) & 255);
    size_t bb = i4 >> 23;
    int n = (int)(i4 & (N_ - 1));
    size_t mcol = bb * N_ + n;
    v4bf h = *(const v4bf*)(h2 + i4);
    float sc = scale2[e], sh = shift2[e];
    float4 o;
    o.x = mask[mcol + 0] ? (float)h[0] * sc + sh : 0.f;
    o.y = mask[mcol + 1] ? (float)h[1] * sc + sh : 0.f;
    o.z = mask[mcol + 2] ? (float)h[2] * sc + sh : 0.f;
    o.w = mask[mcol + 3] ? (float)h[3] * sc + sh : 0.f;
    *(float4*)(out + i4) = o;
}

// ----------------------------------------------------------------- launcher
extern "C" void kernel_launch(void* const* d_in, const int* in_sizes, int n_in,
                              void* d_out, int out_size, void* d_ws, size_t ws_size,
                              hipStream_t stream) {
    const float*         x    = (const float*)d_in[0];
    const unsigned char* mask = (const unsigned char*)d_in[1];
    const float* W1 = (const float*)d_in[2];
    const float* b1 = (const float*)d_in[3];
    const float* g1 = (const float*)d_in[4];
    const float* bt1= (const float*)d_in[5];
    const float* W2 = (const float*)d_in[6];
    const float* b2 = (const float*)d_in[7];
    const float* g2 = (const float*)d_in[8];
    const float* bt2= (const float*)d_in[9];
    float* out = (float*)d_out;

    char* wsb = (char*)d_ws;
    bf16_t* W1b = (bf16_t*)(wsb + 0);            // 128 KB
    bf16_t* W2b = (bf16_t*)(wsb + 131072);       // 128 KB
    float*  st  = (float*)(wsb + 262144);        // stats block
    float *sum1 = st,        *sq1 = st + 256, *sum2 = st + 512, *sq2 = st + 768;
    float *scale1 = st + 1024, *shift1 = st + 1280;
    float *scale2 = st + 1536, *shift2 = st + 1792;
    float *cnt    = st + 2048;
    bf16_t* H1 = (bf16_t*)(wsb + 278528);                    // 134 MB bf16
    bf16_t* H2 = (bf16_t*)(wsb + 278528 + 134217728ull);     // 134 MB bf16

    // zero accumulators (sum/sq/cnt) every launch — ws is not re-poisoned
    (void)hipMemsetAsync(st, 0, 2052 * sizeof(float), stream);

    k_convert_w<<<256, 256, 0, stream>>>(W1, W2, W1b, W2b);
    k_count<<<M_ / 256, 256, 0, stream>>>(mask, cnt);

    dim3 ggrid(M_ / NTILE, E_ / ETILE);          // (8192, 4)
    k_gemm1<<<ggrid, 256, 0, stream>>>(x, mask, W1b, b1, H1, sum1, sq1);
    k_finalize<<<1, 256, 0, stream>>>(sum1, sq1, cnt, g1, bt1, scale1, shift1);
    k_gemm2<<<ggrid, 256, 0, stream>>>(H1, mask, W2b, b2, scale1, shift1, H2, sum2, sq2);
    k_finalize<<<1, 256, 0, stream>>>(sum2, sq2, cnt, g2, bt2, scale2, shift2);
    k_out<<<(M_ * E_ / 4) / 256, 256, 0, stream>>>(H2, mask, scale2, shift2, out);
}
